// Correlation_8873402433912
// MI455X (gfx1250) — matrix operand order repaired and checked by reading, not compiled, not run on hardware
//
#include <hip/hip_runtime.h>

// Problem constants (from reference): B=8, C=256, H=96, W=128, 81 displacement channels
#define B_   8
#define C_   256
#define H_   96
#define W_   128
#define ND_  81
#define HW_  (H_ * W_)

// LDS layout: transposed f16 tiles, [x][c] with padded pitch
#define PITCH   264                 // halfs per row: 256 + 8 pad (528B -> bank stride 4)
#define IN2OFF  (128 * PITCH)       // in1T occupies rows 0..127
#define X2PAD   144                 // in2T columns cover global x in [-4, 139]
#define LDS_HALFS (IN2OFF + X2PAD * PITCH)   // 71808 halfs = 143,616 B

// Extraction scratch (f32), overlaps the in1T region, which is dead once the
// A fragments are register-resident (first dy-loop __syncthreads guarantees it).
#define SCR_PITCH  19                     // floats per tile row (bank-conflict free)
#define SCR_TILE_F (16 * SCR_PITCH)       // 304 floats per 16x16 tile
#define SCR_WAVE_F (2 * SCR_TILE_F)       // 2 tiles per wave (nt = 0,1)

typedef _Float16 v4h  __attribute__((ext_vector_type(4)));
typedef _Float16 v8h  __attribute__((ext_vector_type(8)));
typedef _Float16 v16h __attribute__((ext_vector_type(16)));
typedef float    v4f  __attribute__((ext_vector_type(4)));
typedef float    v8f  __attribute__((ext_vector_type(8)));

static __device__ __forceinline__ v16h cat16(v8h a, v8h b) {
  return __builtin_shufflevector(a, b, 0,1,2,3,4,5,6,7,8,9,10,11,12,13,14,15);
}

static __device__ __forceinline__ v16h loadB(const _Float16* p) {
  return cat16(*(const v8h*)p, *(const v8h*)(p + 16));
}

// Stage one [C=256][W=128] fp32 image row into LDS as f16, transposed to [x][c].
// Coalesced float4 global reads, 4x4 in-register transpose, b64 LDS stores.
static __device__ __forceinline__ void stage_row(_Float16* __restrict__ dst,
                                                 const float* __restrict__ src,
                                                 int tid, bool prefetch_next) {
  const int xb = (tid & 31) * 4;     // lane -> 4 consecutive x (coalesced)
  const int cq = tid >> 5;           // wave id -> c group
#pragma unroll
  for (int j = 0; j < 8; ++j) {
    const int cb = 4 * (cq + 8 * j);
    const float* s0 = src + (size_t)(cb + 0) * HW_ + xb;
    const float* s1 = src + (size_t)(cb + 1) * HW_ + xb;
    const float* s2 = src + (size_t)(cb + 2) * HW_ + xb;
    const float* s3 = src + (size_t)(cb + 3) * HW_ + xb;
    v4f r0 = *(const v4f*)s0;
    v4f r1 = *(const v4f*)s1;
    v4f r2 = *(const v4f*)s2;
    v4f r3 = *(const v4f*)s3;
    if (prefetch_next) __builtin_prefetch(s0 + W_, 0, 0);  // global_prefetch next dy row
#pragma unroll
    for (int i = 0; i < 4; ++i) {
      v4h col = { (_Float16)r0[i], (_Float16)r1[i], (_Float16)r2[i], (_Float16)r3[i] };
      *(v4h*)(dst + (size_t)(xb + i) * PITCH + cb) = col;   // [x][c] transposed
    }
  }
}

// One block (256 threads = 8 wave32) per (b, y). Wave w owns the 16-pixel x-tile x0=16w.
// For each dy: stage in2 row (y+dy-4) into LDS; each wave runs two interleaved
// Gram-matrix WMMA accumulation chains (K=C=256 in 8 steps of 32) with one-step-
// ahead B prefetch, dumps the two 16x16 D tiles to LDS scratch, then stores the 9
// useful diagonals (dx=0..8) with coalesced 64B writes.
extern "C" __global__ __launch_bounds__(256)
void corr81_wmma_f16(const float* __restrict__ in1,
                     const float* __restrict__ in2,
                     float* __restrict__ out) {
  extern __shared__ __attribute__((aligned(16))) _Float16 lds[];
  const int tid  = threadIdx.x;
  const int lane = tid & 31;
  const int wave = tid >> 5;
  const int b = blockIdx.x / H_;
  const int y = blockIdx.x - b * H_;

  // Zero the 16 pad columns of in2T (global x in [-4,0) and [128,140)): spatial zero-pad.
  for (int i = tid; i < 16 * PITCH; i += 256) {
    const int r = i / PITCH, c = i - r * PITCH;
    const int x2 = (r < 4) ? r : (128 + r);
    lds[IN2OFF + x2 * PITCH + c] = (_Float16)0.f;
  }

  // Stage in1 row y once (transposed f16).
  stage_row(lds, in1 + (size_t)b * C_ * HW_ + (size_t)y * W_, tid, false);
  __syncthreads();

  // Per-wave A fragments for all 8 K-steps, held in registers for the whole block.
  // A 16x32 f16 layout: lane m=lane&15; halves 0..7 -> k = 8*hi + [0..7],
  // halves 8..15 -> k = 16 + 8*hi + [0..7]  (hi = lane>=16), global c = 32t + k.
  const int x0 = wave << 4;
  const int hi = lane >> 4;
  const int n  = lane & 15;          // A's m index == B's n index for this lane
  v16h Afrag[8];
#pragma unroll
  for (int t = 0; t < 8; ++t) {
    const _Float16* row = lds + (size_t)(x0 + n) * PITCH;
    const int c1 = 32 * t + 8 * hi;
    Afrag[t] = cat16(*(const v8h*)(row + c1), *(const v8h*)(row + c1 + 16));
  }

  float* scr = (float*)lds + (size_t)wave * SCR_WAVE_F;   // reuses dead in1T region
  const float scale = 1.0f / (float)C_;

  for (int dy = 0; dy < 9; ++dy) {
    const int  ys  = y + dy - 4;
    const bool inb = (ys >= 0) && (ys < H_);
    __syncthreads();                 // all waves done reading previous in2T / scratch
    if (inb)
      stage_row(lds + IN2OFF + 4 * PITCH,
                in2 + (size_t)b * C_ * HW_ + (size_t)ys * W_, tid, ys + 1 < H_);
    __syncthreads();

    if (inb) {
      // B 32x16 f16 layout: lane column n; halves 0..7 -> k = 8*hi + [0..7], halves 8..15 -> k = 16 + 8*hi + [0..7].
      // Tile nt covers in2 global x = x0 - 4 + n + 16*nt  (x2 index = +4).
      const _Float16* b0 = lds + IN2OFF + (size_t)(x0 + n) * PITCH + 8 * hi;
      const _Float16* b1 = b0 + 16 * PITCH;

      // Two independent accumulator chains, interleaved, with one-step-ahead
      // B prefetch: the ds_load_b128 pairs for step t+1 are issued before the
      // two WMMAs of step t, so no WMMA waits cold on LDS.
      v16h B0 = loadB(b0), B1 = loadB(b1);
      v8f acc0 = {0.f, 0.f, 0.f, 0.f, 0.f, 0.f, 0.f, 0.f};
      v8f acc1 = {0.f, 0.f, 0.f, 0.f, 0.f, 0.f, 0.f, 0.f};
#pragma unroll
      for (int t = 0; t < 8; ++t) {
        v16h nB0, nB1;
        if (t < 7) {
          nB0 = loadB(b0 + 32 * (t + 1));
          nB1 = loadB(b1 + 32 * (t + 1));
        }
        acc0 = __builtin_amdgcn_wmma_f32_16x16x32_f16(
            false, Afrag[t], false, B0, (short)0, acc0, false, false);
        acc1 = __builtin_amdgcn_wmma_f32_16x16x32_f16(
            false, Afrag[t], false, B1, (short)0, acc1, false, false);
        if (t < 7) { B0 = nB0; B1 = nB1; }
      }

      // Dump D tiles to LDS scratch: row m = v + 8*hi, col n. Pitch 19 keeps the
      // two half-wave bank sets disjoint (8 rows apart -> +24 banks).
#pragma unroll
      for (int v = 0; v < 8; ++v) {
        scr[(v + 8 * hi) * SCR_PITCH + n]              = acc0[v];
        scr[SCR_TILE_F + (v + 8 * hi) * SCR_PITCH + n] = acc1[v];
      }

      // Branchless diagonal extraction: out(m, dx) = D_{nt}[m][s&15], s = m + dx,
      // nt = s >> 4. 16 consecutive lanes share dx -> 64B-coalesced global stores.
      // 144 values = 4 full wave-wide passes + one 16-lane tail (no exec-mask loop).
      auto emit = [&](int f) {
        const int dx = f >> 4, m = f & 15;
        const int s  = m + dx;
        const float val = scr[(s >> 4) * SCR_TILE_F + m * SCR_PITCH + (s & 15)];
        out[((size_t)(b * ND_ + dy * 9 + dx) * H_ + y) * W_ + x0 + m] = val * scale;
      };
#pragma unroll
      for (int k = 0; k < 4; ++k) emit(lane + 32 * k);
      if (lane < 16) emit(128 + lane);
    } else {
      // dy row out of bounds: correlation against zero padding -> write zeros.
      auto zemit = [&](int f) {
        const int dx = f >> 4, xx = f & 15;
        out[((size_t)(b * ND_ + dy * 9 + dx) * H_ + y) * W_ + x0 + xx] = 0.f;
      };
#pragma unroll
      for (int k = 0; k < 4; ++k) zemit(lane + 32 * k);
      if (lane < 16) zemit(128 + lane);
    }
  }
}

extern "C" void kernel_launch(void* const* d_in, const int* in_sizes, int n_in,
                              void* d_out, int out_size, void* d_ws, size_t ws_size,
                              hipStream_t stream) {
  (void)in_sizes; (void)n_in; (void)d_ws; (void)ws_size; (void)out_size;
  const float* in1 = (const float*)d_in[0];
  const float* in2 = (const float*)d_in[1];
  float* out = (float*)d_out;
  // One block per (b, y): 8*96 = 768 blocks, 256 threads (8 wave32), ~140 KB dynamic LDS.
  corr81_wmma_f16<<<dim3(B_ * H_), dim3(256), (size_t)LDS_HALFS * sizeof(_Float16),
                    stream>>>(in1, in2, out);
}
